// FLowLD_82660940579152
// MI455X (gfx1250) — 8-bit matrix operand order repaired and checked by reading, compiled, not run on hardware
//
#include <hip/hip_runtime.h>
#include <hip/hip_bf16.h>

#define N_LEVELS 100
#define C_CH 3
#define F_FEAT 32
#define D_DIM 1024
#define COMP 4
#define B_BATCH 16
#define S_SEQ 64
#define K_PAD 4096
#define M_ROWS (B_BATCH * S_SEQ * C_CH)
#define NT_PER_WAVE 4

typedef __attribute__((ext_vector_type(8))) int v8i;
typedef __attribute__((ext_vector_type(4))) int v4i;

__global__ void k_quantize(const float* __restrict__ samples,
                           unsigned char* __restrict__ idx) {
  int i = blockIdx.x * 256 + threadIdx.x;
  float x = samples[i];
  float q = rintf((x + 1.0f) * 0.5f * 99.0f);
  q = fminf(fmaxf(q, 0.0f), 99.0f);
  idx[i] = (unsigned char)(int)q;
}

__global__ void k_build_wt(const float* __restrict__ value_w,
                           const float* __restrict__ feat_w,
                           signed char* __restrict__ Wt) {
  int k = blockIdx.x * 256 + threadIdx.x;
  int d = blockIdx.y;
  int f = k >> 7;
  int l = k & 127;
  float p = 0.0f;
  if (l < N_LEVELS)
    p = feat_w[f * D_DIM + d] * value_w[l * D_DIM + d];
  Wt[(long)d * K_PAD + k] = (signed char)(int)p;
}

__global__ void k_gemm_wmma(const unsigned char* __restrict__ idx,
                            const signed char* __restrict__ Wt,
                            float* __restrict__ h) {
  __shared__ int lidx_i[128];
  const int mtile = blockIdx.y;
  const int lane  = threadIdx.x & 31;
  const int wave  = threadIdx.x >> 5;

  lidx_i[threadIdx.x] =
      ((const int*)(idx + (long)mtile * 16 * F_FEAT))[threadIdx.x];
  __syncthreads();

  const int m = lane & 15;
  const int g = lane >> 4;
  const int dcol = (blockIdx.x * 4 + wave) * (16 * NT_PER_WAVE) + (lane & 15);
  const signed char* brows[NT_PER_WAVE];
#pragma unroll
  for (int j = 0; j < NT_PER_WAVE; ++j)
    brows[j] = Wt + (long)(dcol + 16 * j) * K_PAD;

  const unsigned char* myidx = ((const unsigned char*)lidx_i) + m * F_FEAT;
  v4i wlo = *(const v4i*)(myidx);
  v4i whi = *(const v4i*)(myidx + 16);
  int wq[8] = {wlo[0], wlo[1], wlo[2], wlo[3], whi[0], whi[1], whi[2], whi[3]};

  const int gshift = g << 2;
  v8i acc[NT_PER_WAVE];
#pragma unroll
  for (int j = 0; j < NT_PER_WAVE; ++j) acc[j] = (v8i){0, 0, 0, 0, 0, 0, 0, 0};

#pragma unroll
  for (int f = 0; f < F_FEAT; ++f) {
    const int idxv = (wq[f >> 2] >> ((f & 3) * 8)) & 255;
    const unsigned int hot = 1u << ((idxv & 3) << 3);
    const int tg = (idxv >> 2) - gshift;

    const int koff = f * 128 + g * 16;

    v4i xb[4 * NT_PER_WAVE];
#pragma unroll
    for (int j = 0; j < NT_PER_WAVE; ++j) {
      xb[j * 4 + 0] = *(const v4i*)(brows[j] + koff);
      xb[j * 4 + 1] = *(const v4i*)(brows[j] + koff + 32);
      xb[j * 4 + 2] = *(const v4i*)(brows[j] + koff + 64);
      xb[j * 4 + 3] = *(const v4i*)(brows[j] + koff + 96);
    }

    v8i alo, ahi;
#pragma unroll
    for (int v = 0; v < 8; ++v) {
      const int grp = (v & 3) + (v >> 2) * 8;
      alo[v] = (tg == grp) ? (int)hot : 0;
      ahi[v] = (tg == grp + 16) ? (int)hot : 0;
    }

#pragma unroll
    for (int j = 0; j < NT_PER_WAVE; ++j) {
      v4i x0 = xb[j * 4 + 0], x1 = xb[j * 4 + 1];
      v8i b;
      b[0] = x0[0]; b[1] = x0[1]; b[2] = x0[2]; b[3] = x0[3];
      b[4] = x1[0]; b[5] = x1[1]; b[6] = x1[2]; b[7] = x1[3];
      acc[j] = __builtin_amdgcn_wmma_i32_16x16x64_iu8(false, alo, true, b,
                                                      acc[j], false, false);
    }
#pragma unroll
    for (int j = 0; j < NT_PER_WAVE; ++j) {
      v4i x0 = xb[j * 4 + 2], x1 = xb[j * 4 + 3];
      v8i b;
      b[0] = x0[0]; b[1] = x0[1]; b[2] = x0[2]; b[3] = x0[3];
      b[4] = x1[0]; b[5] = x1[1]; b[6] = x1[2]; b[7] = x1[3];
      acc[j] = __builtin_amdgcn_wmma_i32_16x16x64_iu8(false, ahi, true, b,
                                                      acc[j], false, false);
    }
  }

#pragma unroll
  for (int j = 0; j < NT_PER_WAVE; ++j) {
#pragma unroll
    for (int r = 0; r < 8; ++r) {
      int row = mtile * 16 + r + 8 * g;
      h[(long)row * D_DIM + dcol + 16 * j] = (float)acc[j][r];
    }
  }
}

__global__ void k_combine(const float* __restrict__ h,
                          const float* __restrict__ comp_w,
                          float* __restrict__ M, float* __restrict__ H) {
  int d = blockIdx.x * 256 + threadIdx.x;
  int b = blockIdx.y;
  float cw0 = comp_w[0 * D_DIM + d], cw1 = comp_w[1 * D_DIM + d];
  float cw2 = comp_w[2 * D_DIM + d], cw3 = comp_w[3 * D_DIM + d];
  const float w0  = cw0 + cw1 + cw2;
  const float w1a = cw3, w1b = cw0 + cw1;
  const float w2a = cw2 + cw3, w2b = cw0;
  const float w3  = cw1 + cw2 + cw3;
  float a0 = 0.f, a1 = 0.f, a2 = 0.f, a3 = 0.f, ah = 0.f;
  for (int s = 0; s < S_SEQ; ++s) {
    long base = ((long)(b * S_SEQ + s) * C_CH) * D_DIM + d;
    float h0 = h[base], h1 = h[base + D_DIM], h2 = h[base + 2 * D_DIM];
    float m0 = h0 * w0;
    float m1 = h0 * w1a + h1 * w1b;
    float m2 = h1 * w2a + h2 * w2b;
    float m3 = h2 * w3;
    float t  = 1.0f / (1.0f + __expf(-(m2 + m3)));
    float hs = m0 * (1.0f - t) + t * m1;
    a0 += m0; a1 += m1; a2 += m2; a3 += m3; ah += hs;
  }
  M[((b * COMP + 0) * D_DIM) + d] = a0;
  M[((b * COMP + 1) * D_DIM) + d] = a1;
  M[((b * COMP + 2) * D_DIM) + d] = a2;
  M[((b * COMP + 3) * D_DIM) + d] = a3;
  H[b * D_DIM + d] = ah;
}

__global__ void k_finalize(const float* __restrict__ M,
                           const float* __restrict__ H,
                           float* __restrict__ out) {
  int i = blockIdx.x * 256 + threadIdx.x;
  int b = i >> 12;
  int pd = i & 4095;
  int d = pd & (D_DIM - 1);
  float v = M[b * (COMP * D_DIM) + pd] + (b ? H[(b - 1) * D_DIM + d] : 0.0f);
  out[i] = (v > 0.f) ? 1.f : ((v < 0.f) ? -1.f : 0.f);
}

extern "C" void kernel_launch(void* const* d_in, const int* in_sizes, int n_in,
                              void* d_out, int out_size, void* d_ws,
                              size_t ws_size, hipStream_t stream) {
  const float* samples = (const float*)d_in[0];
  const float* value_w = (const float*)d_in[1];
  const float* feat_w  = (const float*)d_in[2];
  const float* comp_w  = (const float*)d_in[3];
  float* out = (float*)d_out;

  char* ws = (char*)d_ws;
  unsigned char* idx = (unsigned char*)ws;
  signed char*   Wt  = (signed char*)(ws + 98304);
  float* h = (float*)(ws + 98304 + 4194304);
  float* M = (float*)(ws + 98304 + 4194304 + 12582912);
  float* H = (float*)(ws + 98304 + 4194304 + 12582912 + 262144);

  k_quantize<<<dim3(M_ROWS * F_FEAT / 256), 256, 0, stream>>>(samples, idx);
  k_build_wt<<<dim3(K_PAD / 256, D_DIM), 256, 0, stream>>>(value_w, feat_w, Wt);
  k_gemm_wmma<<<dim3(D_DIM / 16 / (4 * NT_PER_WAVE), M_ROWS / 16), 128, 0,
                stream>>>(idx, Wt, h);
  k_combine<<<dim3(D_DIM / 256, B_BATCH), 256, 0, stream>>>(h, comp_w, M, H);
  k_finalize<<<dim3(B_BATCH * COMP * D_DIM / 256), 256, 0, stream>>>(M, H, out);
}
